// EncoderBlock_10204842295863
// MI455X (gfx1250) — matrix operand order repaired and checked by reading, not compiled, not run on hardware
//
#include <hip/hip_runtime.h>
#include <cstdint>
#include <cstddef>

// ---------------- problem sizes (fixed by the reference) ----------------
#define Bq   4
#define Tq   1024
#define Cq   1024
#define Hq   16
#define HDq  64
#define FFq  4096
#define EPSq 1e-5f

// ---------------- CDNA5 types / WMMA ----------------
typedef __attribute__((ext_vector_type(16))) __bf16   v16bf;
typedef __attribute__((ext_vector_type(8)))  float    v8f;
typedef __attribute__((ext_vector_type(4)))  int      i4;
typedef __attribute__((ext_vector_type(4)))  float    f4;
typedef __attribute__((ext_vector_type(4)))  unsigned u32x4;
typedef __attribute__((ext_vector_type(8)))  int      i32x8;

#define WMMA_BF16(a, b, c) \
  __builtin_amdgcn_wmma_f32_16x16x32_bf16(false, (a), false, (b), (short)0, (c), false, false)

// TDM writes LDS behind the compiler's back (the builtin carries no pointer
// operand). A plain "memory" clobber is NOT enough: if the LDS global is never
// stored and never escapes, GlobalOpt folds its loads to undef. Passing the
// pointer as an asm operand makes it escape, so the asm may write through it,
// forcing the ds_load fragments to be materialized and ordered after the fence.
#define LDS_FENCE(p) asm volatile("" :: "v"((const void*)(p)) : "memory")

union FragU  { v16bf v; i4 q[2]; };
union PFrag  { v16bf v; unsigned u[8]; };
union LVU    { i4 q[4]; __bf16 h[32]; };

// A-fragment (16xK=32, bf16) from a row-major LDS tile [rows][stride]
// Layout used here: lane<16 holds k 0..15, lane>=16 holds k 16..31 (contiguous), row = lane%16.
__device__ __forceinline__ v16bf lds_frag_a(const __bf16* base, int row, int stride, int kbase) {
  const int lane = threadIdx.x & 31;
  const __bf16* p = base + (row + (lane & 15)) * stride + kbase + ((lane < 16) ? 0 : 16);
  FragU u;
  u.q[0] = *(const i4*)(p);
  u.q[1] = *(const i4*)(p + 8);
  return u.v;
}

// B-fragment (K=32 x 16, bf16) from a row-major LDS tile [N][stride] (B stored N x K).
// Layout used here: col = lane%16; lane<16 holds k 0..15, lane>=16 holds k 16..31 (contiguous).
__device__ __forceinline__ v16bf lds_frag_b(const __bf16* base, int col, int stride, int kbase) {
  const int lane = threadIdx.x & 31;
  const __bf16* p = base + (col + (lane & 15)) * stride + kbase + ((lane < 16) ? 0 : 16);
  FragU u;
  u.q[0] = *(const i4*)(p);
  u.q[1] = *(const i4*)(p + 8);
  return u.v;
}

__device__ __forceinline__ unsigned pack_bf16(float a, float b) {
  union { __bf16 h[2]; unsigned u; } v;
  v.h[0] = (__bf16)a; v.h[1] = (__bf16)b;
  return v.u;
}

// ---------------- TDM: 2D bf16 tile load -> LDS (with LDS row padding) ----------------
// tile_d0 elems per row (2B each), tile_d1 rows, row stride = stride0 elems.
// pad_int: pad_interval code (3 = every 16 DWORDs, 4 = every 32 DWORDs); pad = 4 DWORDs.
// D# bit layout per CDNA5 ISA ch.8: group0 {count, lds_addr, global_addr, type=2},
// group1 {data_size=1(2B), pad_enable, pad_interval, pad_amount=3(4DW),
//         tensor_dim0/1, tile_dim0/1, tensor_dim0_stride}.
__device__ __forceinline__ void tdm_load_2d(unsigned lds_addr, const void* gptr,
                                            int tile_d0, int tile_d1, long stride0,
                                            int pad_int) {
  const unsigned long long ga = (unsigned long long)gptr;
  u32x4 g0;
  g0[0] = 1u;                                                   // count=1, no gather
  g0[1] = lds_addr;                                             // [63:32] LDS byte address
  g0[2] = (unsigned)ga;                                         // [95:64] global addr lo
  g0[3] = (unsigned)((ga >> 32) & 0x01FFFFFFull) | (2u << 30);  // [120:96] hi, type=2

  const unsigned td0 = (unsigned)tile_d0;  // tensor dims = tile dims (no OOB clipping)
  const unsigned td1 = (unsigned)tile_d1;
  const unsigned long long s0 = (unsigned long long)stride0;
  i32x8 g1;
  g1[0] = (int)((1u << 16) | (1u << 20) | ((unsigned)pad_int << 22) | (3u << 25));
  g1[1] = (int)((td0 & 0xFFFFu) << 16);                    // [47:32]=0, [63:48]=td0.lo
  g1[2] = (int)((td0 >> 16) | ((td1 & 0xFFFFu) << 16));    // td0.hi | td1.lo
  g1[3] = (int)((td1 >> 16) | ((unsigned)tile_d0 << 16));  // td1.hi | tile_dim0
  g1[4] = (int)((unsigned)tile_d1 & 0xFFFFu);              // tile_dim1, tile_dim2=0
  g1[5] = (int)(unsigned)(s0 & 0xFFFFFFFFull);             // stride0[31:0]
  g1[6] = (int)(unsigned)((s0 >> 32) & 0xFFFFull);         // stride0[47:32], stride1.lo=0
  g1[7] = 0;

  const i4 z4 = {0, 0, 0, 0};
#if __clang_major__ >= 23
  const i32x8 z8 = {0, 0, 0, 0, 0, 0, 0, 0};
  __builtin_amdgcn_tensor_load_to_lds(g0, g1, z4, z4, z8, 0);
#else
  __builtin_amdgcn_tensor_load_to_lds(g0, g1, z4, z4, 0);
#endif
}

// ---------------- fp32 -> bf16 convert ----------------
__global__ __launch_bounds__(256) void cvt_f32_bf16(const float* __restrict__ in,
                                                    __bf16* __restrict__ out, int n) {
  int i = blockIdx.x * 256 + threadIdx.x;
  if (i < n) out[i] = (__bf16)in[i];
}

// ---------------- WMMA GEMM: C = A(MxK) * B(NxK)^T (+bias)(+relu) ----------------
// 128x64 tile, 4 waves, TDM-staged double-buffered LDS (one barrier per K-slab).
__global__ __launch_bounds__(128) void gemm_bf16(
    const __bf16* __restrict__ A, int lda, long sA0, long sA1,
    const __bf16* __restrict__ Bm, int ldb, long sB0, long sB1,
    float* Cf, __bf16* Cbf, int ldc, long sC0, long sC1,
    const float* __restrict__ bias, int M, int N, int K, int nb1, int relu)
{
  const int z  = blockIdx.z;
  const int i0 = z / nb1, i1 = z % nb1;
  A  += (long)i0 * sA0 + (long)i1 * sA1;
  Bm += (long)i0 * sB0 + (long)i1 * sB1;
  const long coff = (long)i0 * sC0 + (long)i1 * sC1;
  if (Cf)  Cf  += coff;
  if (Cbf) Cbf += coff;

  constexpr int AROWS = 128, BROWS = 64, STR = 40;       // 80B LDS rows (16B aligned)
  constexpr int BUFE  = (AROWS + BROWS) * STR;           // elements per buffer
  constexpr unsigned ABYTES = AROWS * STR * 2;
  constexpr unsigned BUFB   = BUFE * 2;
  __shared__ __bf16 smem[2 * BUFE];                      // 30 KB: [buf][A(128x32) | B(64x32)]

  const int tid  = threadIdx.x;
  const int wave = tid >> 5;
  const int lane = tid & 31;
  const int rowBase = blockIdx.y * 128;
  const int colBase = blockIdx.x * 64;

  const int nslab = K / 32;
  auto stage = [&](int slab, int p) {
    if (wave == 0) {
      tdm_load_2d((unsigned)p * BUFB,
                  A + (long)rowBase * lda + slab * 32, 32, AROWS, (long)lda, 3);
      tdm_load_2d((unsigned)p * BUFB + ABYTES,
                  Bm + (long)colBase * ldb + slab * 32, 32, BROWS, (long)ldb, 3);
    }
  };

  v8f acc[2][4] = {};   // 2 row strips (wave*32 + {0,16}) x 4 col tiles
  stage(0, 0);
  for (int i = 0; i < nslab; ++i) {
    const int p = i & 1;
    if (wave == 0) __builtin_amdgcn_s_wait_tensorcnt(0);
    __syncthreads();                       // buf p ready for everyone; buf p^1 free
    LDS_FENCE(smem);                       // escape + clobber: TDM-written LDS is live
    if (i + 1 < nslab) stage(i + 1, p ^ 1);

    const __bf16* bufA = smem + p * BUFE;
    const __bf16* bufB = bufA + AROWS * STR;
    v16bf a0 = lds_frag_a(bufA, wave * 32,      STR, 0);
    v16bf a1 = lds_frag_a(bufA, wave * 32 + 16, STR, 0);
#pragma unroll
    for (int j = 0; j < 4; ++j) {
      v16bf bf = lds_frag_b(bufB, j * 16, STR, 0);
      acc[0][j] = WMMA_BF16(a0, bf, acc[0][j]);
      acc[1][j] = WMMA_BF16(a1, bf, acc[1][j]);
    }
  }

  // Epilogue: D-frag mapping: col = lane%16, row = r + (lane>=16 ? 8 : 0)
#pragma unroll
  for (int s = 0; s < 2; ++s) {
    const int rbase = rowBase + wave * 32 + s * 16 + ((lane < 16) ? 0 : 8);
#pragma unroll
    for (int j = 0; j < 4; ++j) {
      const int gc = colBase + j * 16 + (lane & 15);
      const float bv = bias ? bias[gc] : 0.f;
#pragma unroll
      for (int r = 0; r < 8; ++r) {
        float v = acc[s][j][r] + bv;
        if (relu) v = fmaxf(v, 0.f);
        const long idx = (long)(rbase + r) * ldc + gc;
        if (Cf)  Cf[idx]  = v;
        if (Cbf) Cbf[idx] = (__bf16)v;
      }
    }
  }
}

// ---------------- flash attention ----------------
// grid: (T/64, B*H); 128 threads. Q/K/V bf16 laid out [(b*H+h)*T + t][64].
// Each wave: 16 queries. S^T = K_tile * Q^T so softmax rows live on the lane dim pair (L, L^16).
// K tile staged by TDM (overlaps with the manual V-transpose staging).
__global__ __launch_bounds__(128) void flash_attn(
    const __bf16* __restrict__ Q, const __bf16* __restrict__ K,
    const __bf16* __restrict__ V, float* __restrict__ ctx)
{
  const int bh = blockIdx.y;
  const int b  = bh >> 4, h = bh & 15;
  const int qbase = blockIdx.x * 64;
  const int tid = threadIdx.x, wave = tid >> 5, lane = tid & 31;

  constexpr int TILE_E = 64 * 72;                 // elements per 64x64(+pad) tile
  constexpr unsigned SK_BYTE = TILE_E * 2;        // sK LDS byte offset (segment base 0)
  __shared__ __bf16 smem[3 * TILE_E];             // [ sQ | sK | sVt ]
  __bf16* sQ  = smem;
  __bf16* sK  = smem + TILE_E;
  __bf16* sVt = smem + 2 * TILE_E;                // V transposed: [d][key]

  const __bf16* Qp = Q + ((long)bh * Tq + qbase) * HDq;
  const __bf16* Kp = K + (long)bh * Tq * HDq;
  const __bf16* Vp = V + (long)bh * Tq * HDq;

  // stage the 64x64 Q tile
  {
    const int r = tid >> 1, cseg = (tid & 1) * 32;
    const __bf16* g = Qp + r * HDq + cseg;
#pragma unroll
    for (int j = 0; j < 4; ++j)
      *(i4*)(&sQ[r * 72 + cseg + j * 8]) = *(const i4*)(g + j * 8);
  }
  __syncthreads();

  // loop-invariant Q B-fragments (queries on the lane dim, d contiguous)
  v16bf qb0 = lds_frag_b(sQ, wave * 16, 72, 0);
  v16bf qb1 = lds_frag_b(sQ, wave * 16, 72, 32);

  float m = -1e30f, l = 0.f;
  v8f o[4] = {};                    // O^T: d (4 strips of 16) x 16 queries
  const float scale = 0.125f;       // HD^-0.5

  for (int kt = 0; kt < Tq; kt += 64) {
    __syncthreads();                // previous iteration's fragment reads complete
    // K tile via TDM: 64 rows of 64 elems (128B = 32 DW), pad 4 DW -> stride 72
    if (wave == 0)
      tdm_load_2d(SK_BYTE, Kp + (long)kt * HDq, 64, 64, (long)HDq, 4);
    // V tile: manual load + transposed store (TDM cannot transpose)
    {
      const int r = tid >> 1, cseg = (tid & 1) * 32;
      const __bf16* gv = Vp + (long)(kt + r) * HDq + cseg;
      if (kt + 64 < Tq)
        __builtin_prefetch(gv + 64 * HDq, 0, 1);   // -> global_prefetch_b8
      LVU lv;
#pragma unroll
      for (int j = 0; j < 4; ++j) lv.q[j] = *(const i4*)(gv + j * 8);
#pragma unroll
      for (int i = 0; i < 32; ++i) sVt[(cseg + i) * 72 + r] = lv.h[i];
    }
    if (wave == 0) __builtin_amdgcn_s_wait_tensorcnt(0);
    __syncthreads();
    LDS_FENCE(smem);                // escape + clobber: TDM-written sK is live

#pragma unroll
    for (int step = 0; step < 2; ++step) {
      const int kb = step * 32;
      // S^T tiles: s0 -> keys kb..kb+15, s1 -> keys kb+16..kb+31 (x 16 queries)
      v8f s0 = {}, s1 = {};
      {
        v16bf ka = lds_frag_a(sK, kb, 72, 0);
        s0 = WMMA_BF16(ka, qb0, s0);
        v16bf kb2 = lds_frag_a(sK, kb, 72, 32);
        s0 = WMMA_BF16(kb2, qb1, s0);
        v16bf kc = lds_frag_a(sK, kb + 16, 72, 0);
        s1 = WMMA_BF16(kc, qb0, s1);
        v16bf kd = lds_frag_a(sK, kb + 16, 72, 32);
        s1 = WMMA_BF16(kd, qb1, s1);
      }
      // online softmax over 32 keys; query = lane%16, shared by lanes (L, L^16)
      float tmax = -1e30f;
#pragma unroll
      for (int r = 0; r < 8; ++r) {
        s0[r] *= scale; s1[r] *= scale;
        tmax = fmaxf(tmax, fmaxf(s0[r], s1[r]));
      }
      tmax = fmaxf(tmax, __shfl_xor(tmax, 16, 32));
      const float mnew  = fmaxf(m, tmax);
      const float alpha = __expf(m - mnew);
      float p0[8], p1[8], lsum = 0.f;
#pragma unroll
      for (int r = 0; r < 8; ++r) {
        p0[r] = __expf(s0[r] - mnew);
        p1[r] = __expf(s1[r] - mnew);
        lsum += p0[r] + p1[r];
      }
      lsum += __shfl_xor(lsum, 16, 32);
      l = l * alpha + lsum;
      m = mnew;
#pragma unroll
      for (int j = 0; j < 4; ++j)
#pragma unroll
        for (int r = 0; r < 8; ++r) o[j][r] *= alpha;

      // Repack P (two D-frags) into one B-frag (32 keys x 16 queries) via lane-pair shuffles
      unsigned pk0[4], pk1[4], x0[4], x1[4];
#pragma unroll
      for (int i = 0; i < 4; ++i) {
        pk0[i] = pack_bf16(p0[2 * i], p0[2 * i + 1]);
        pk1[i] = pack_bf16(p1[2 * i], p1[2 * i + 1]);
      }
#pragma unroll
      for (int i = 0; i < 4; ++i) {
        x0[i] = __shfl_xor(pk0[i], 16, 32);
        x1[i] = __shfl_xor(pk1[i], 16, 32);
      }
      PFrag pb;
#pragma unroll
      for (int i = 0; i < 4; ++i) {
        pb.u[i]     = (lane < 16) ? pk0[i] : x1[i];
        pb.u[i + 4] = (lane < 16) ? x0[i]  : pk1[i];
      }

      // O^T += V^T(16d x 32keys) * P^T(32keys x 16q)
#pragma unroll
      for (int j = 0; j < 4; ++j) {
        v16bf va = lds_frag_a(sVt, j * 16, 72, kb);
        o[j] = WMMA_BF16(va, pb.v, o[j]);
      }
    }
  }

  // epilogue: divide by l, write ctx[b][q][h*64 + d] (d contiguous per lane)
  const float inv = 1.f / l;
  const int q = qbase + wave * 16 + (lane & 15);
  float* cp = ctx + ((long)b * Tq + q) * Cq + h * HDq;
#pragma unroll
  for (int j = 0; j < 4; ++j) {
    const int dbase = j * 16 + ((lane < 16) ? 0 : 8);
    f4 v0, v1;
#pragma unroll
    for (int r = 0; r < 4; ++r) { v0[r] = o[j][r] * inv; v1[r] = o[j][r + 4] * inv; }
    *(f4*)(cp + dbase)     = v0;
    *(f4*)(cp + dbase + 4) = v1;
  }
}

// ---------------- residual add + LayerNorm (row = one workgroup of 256) ----------------
__global__ __launch_bounds__(256) void add_ln(
    const float* __restrict__ a, const float* __restrict__ bsrc,
    const float* __restrict__ g, const float* __restrict__ be,
    float* __restrict__ outf, __bf16* outb)
{
  const int row = blockIdx.x;
  const int tid = threadIdx.x;
  const long base = (long)row * Cq;

  float x[4]; float s = 0.f;
#pragma unroll
  for (int i = 0; i < 4; ++i) {
    const int c = tid + i * 256;
    x[i] = a[base + c] + bsrc[base + c];
    s += x[i];
  }
  __shared__ float red[8];
#pragma unroll
  for (int off = 16; off > 0; off >>= 1) s += __shfl_xor(s, off, 32);
  if ((tid & 31) == 0) red[tid >> 5] = s;
  __syncthreads();
  float mu = 0.f;
#pragma unroll
  for (int w = 0; w < 8; ++w) mu += red[w];
  mu *= (1.f / (float)Cq);

  float vs = 0.f;
#pragma unroll
  for (int i = 0; i < 4; ++i) { const float d = x[i] - mu; vs += d * d; }
  __syncthreads();
#pragma unroll
  for (int off = 16; off > 0; off >>= 1) vs += __shfl_xor(vs, off, 32);
  if ((tid & 31) == 0) red[tid >> 5] = vs;
  __syncthreads();
  float var = 0.f;
#pragma unroll
  for (int w = 0; w < 8; ++w) var += red[w];
  var *= (1.f / (float)Cq);
  const float rstd = rsqrtf(var + EPSq);

#pragma unroll
  for (int i = 0; i < 4; ++i) {
    const int c = tid + i * 256;
    const float y = (x[i] - mu) * rstd * g[c] + be[c];
    outf[base + c] = y;
    if (outb) outb[base + c] = (__bf16)y;
  }
}

// ---------------- workspace layout (bytes, with phase reuse) ----------------
static constexpr size_t SZ_WQKV = (size_t)3 * Hq * HDq * HDq * 2;     // 384 KB
static constexpr size_t SZ_W1   = (size_t)FFq * Cq * 2;               // 8 MB
static constexpr size_t SZ_W2   = (size_t)Cq * FFq * 2;               // 8 MB
static constexpr size_t SZ_XBF  = (size_t)Bq * Tq * Cq * 2;           // 8 MB
static constexpr size_t SZ_QKV1 = (size_t)Bq * Hq * Tq * HDq * 2;     // 8 MB each
static constexpr size_t SZ_F32  = (size_t)Bq * Tq * Cq * 4;           // 16 MB

static constexpr size_t OFF_WQKV = 0;
static constexpr size_t OFF_W1   = OFF_WQKV + SZ_WQKV;
static constexpr size_t OFF_W2   = OFF_W1 + SZ_W1;
static constexpr size_t OFF_HF   = OFF_W2 + SZ_W2;                    // h (f32), persists
static constexpr size_t OFF_HBF  = OFF_HF + SZ_F32;                   // h (bf16)
static constexpr size_t OFF_CTX  = OFF_HBF + SZ_XBF;                  // ctx f32; reused as ff2 f32
static constexpr size_t OFF_A    = OFF_CTX + SZ_F32;                  // Xbf+QKV; reused as ff1 bf16
static constexpr size_t OFF_XBF  = OFF_A;
static constexpr size_t OFF_Q    = OFF_A + SZ_XBF;
static constexpr size_t OFF_K    = OFF_Q + SZ_QKV1;
static constexpr size_t OFF_V    = OFF_K + SZ_QKV1;
static constexpr size_t OFF_FF1  = OFF_A;                             // 32 MB bf16

extern "C" void kernel_launch(void* const* d_in, const int* in_sizes, int n_in,
                              void* d_out, int out_size, void* d_ws, size_t ws_size,
                              hipStream_t stream)
{
  (void)in_sizes; (void)n_in; (void)out_size; (void)ws_size;

  const float* X   = (const float*)d_in[0];
  const float* Wqw = (const float*)d_in[1];
  const float* Wkw = (const float*)d_in[2];
  const float* Wvw = (const float*)d_in[3];
  const float* g1  = (const float*)d_in[4];
  const float* be1 = (const float*)d_in[5];
  const float* W1  = (const float*)d_in[6];
  const float* bf1 = (const float*)d_in[7];
  const float* W2  = (const float*)d_in[8];
  const float* bf2 = (const float*)d_in[9];
  const float* g2  = (const float*)d_in[10];
  const float* be2 = (const float*)d_in[11];

  char* ws = (char*)d_ws;
  __bf16* wqkv = (__bf16*)(ws + OFF_WQKV);
  __bf16* w1b  = (__bf16*)(ws + OFF_W1);
  __bf16* w2b  = (__bf16*)(ws + OFF_W2);
  __bf16* xbf  = (__bf16*)(ws + OFF_XBF);
  __bf16* qb   = (__bf16*)(ws + OFF_Q);
  __bf16* kb   = (__bf16*)(ws + OFF_K);
  __bf16* vb   = (__bf16*)(ws + OFF_V);
  float*  ctx  = (float*)(ws + OFF_CTX);
  float*  hf   = (float*)(ws + OFF_HF);
  __bf16* hbf  = (__bf16*)(ws + OFF_HBF);
  __bf16* ff1  = (__bf16*)(ws + OFF_FF1);
  float*  ff2  = (float*)(ws + OFF_CTX);
  float*  out  = (float*)d_out;

  const int nX = Bq * Tq * Cq;
  const int nW = Hq * HDq * HDq;
  const int nF = FFq * Cq;

  // Phase 0: convert inputs / weights to bf16
  cvt_f32_bf16<<<(nX + 255) / 256, 256, 0, stream>>>(X,   xbf, nX);
  cvt_f32_bf16<<<(nW + 255) / 256, 256, 0, stream>>>(Wqw, wqkv,          nW);
  cvt_f32_bf16<<<(nW + 255) / 256, 256, 0, stream>>>(Wkw, wqkv + nW,     nW);
  cvt_f32_bf16<<<(nW + 255) / 256, 256, 0, stream>>>(Wvw, wqkv + 2 * nW, nW);
  cvt_f32_bf16<<<(nF + 255) / 256, 256, 0, stream>>>(W1,  w1b, nF);
  cvt_f32_bf16<<<(nF + 255) / 256, 256, 0, stream>>>(W2,  w2b, nF);

  // Phase 1: QKV projections (batched over (b,h): A = X head slice, B = W[h] 64x64)
  {
    dim3 grid(1, Tq / 128, Bq * Hq);
    __bf16* outs[3] = {qb, kb, vb};
    for (int mtx = 0; mtx < 3; ++mtx) {
      gemm_bf16<<<grid, 128, 0, stream>>>(
          xbf, Cq, (long)Tq * Cq, (long)HDq,
          wqkv + (size_t)mtx * nW, HDq, 0L, (long)HDq * HDq,
          nullptr, outs[mtx], HDq, (long)Hq * Tq * HDq, (long)Tq * HDq,
          nullptr, Tq, HDq, HDq, Hq, 0);
    }
  }

  // Phase 2: flash attention -> ctx (f32, [B,T,C])
  flash_attn<<<dim3(Tq / 64, Bq * Hq), 128, 0, stream>>>(qb, kb, vb, ctx);

  // Phase 3: h = LN1(X + ctx); also emit h in bf16 for the FFN GEMM
  add_ln<<<Bq * Tq, 256, 0, stream>>>(X, ctx, g1, be1, hf, hbf);

  // Phase 4: FFN1 = relu(h @ W1^T + bf1) -> bf16 (reuses region A)
  gemm_bf16<<<dim3(FFq / 64, (Bq * Tq) / 128, 1), 128, 0, stream>>>(
      hbf, Cq, 0L, 0L,
      w1b, Cq, 0L, 0L,
      nullptr, ff1, FFq, 0L, 0L,
      bf1, Bq * Tq, FFq, Cq, 1, 1);

  // Phase 5: FFN2 = ff1 @ W2^T + bf2 -> f32 (reuses ctx region)
  gemm_bf16<<<dim3(Cq / 64, (Bq * Tq) / 128, 1), 128, 0, stream>>>(
      ff1, FFq, 0L, 0L,
      w2b, FFq, 0L, 0L,
      ff2, nullptr, Cq, 0L, 0L,
      bf2, Bq * Tq, Cq, FFq, 1, 0);

  // Phase 6: out = LN2(h + ff2)
  add_ln<<<Bq * Tq, 256, 0, stream>>>(hf, ff2, g2, be2, out, nullptr);
}
